// SkinningField_61959198212439
// MI455X (gfx1250) — matrix operand order repaired and checked by reading, not compiled, not run on hardware
//
#include <hip/hip_runtime.h>
#include <hip/hip_bf16.h>

typedef __attribute__((ext_vector_type(16))) _Float16 v16h;
typedef __attribute__((ext_vector_type(8)))  _Float16 v8h;
typedef __attribute__((ext_vector_type(8)))  float    v8f;

#define NPTS 524288
#define WSTR 136   // half stride for 128-wide weight/act rows (272B, padded vs 64 banks)
#define LSTR 68    // float stride for logit/T rows (272B -> aliases act rows exactly)
#define TSTR 72    // half stride for tfs rows
#define SMEM_BYTES 127520
#define L2E 1.4426950408889634f
#define LN2 0.6931471805599453f

// ---- raw transcendental helpers (skip denorm fixups; args are range-safe) ----
__device__ __forceinline__ float fexp2_(float x) {
#if __has_builtin(__builtin_amdgcn_exp2f)
  return __builtin_amdgcn_exp2f(x);
#else
  return __expf(x * LN2);
#endif
}
__device__ __forceinline__ float flog2_(float x) {
#if __has_builtin(__builtin_amdgcn_logf)
  return __builtin_amdgcn_logf(x);
#else
  return __logf(x) * L2E;
#endif
}
__device__ __forceinline__ float frcp_(float x) { return __builtin_amdgcn_rcpf(x); }
__device__ __forceinline__ float frsq_(float x) {
#if __has_builtin(__builtin_amdgcn_rsqf)
  return __builtin_amdgcn_rsqf(x);
#else
  return rsqrtf(x);
#endif
}

// softplus(x) = max(x,0) + ln2 * log2(1 + 2^(-|x|*log2e))   (branchless, stable)
__device__ __forceinline__ float sp_(float x) {
  float t = flog2_(1.f + fexp2_(-fabsf(x) * L2E)) * LN2;
  return fmaxf(x, 0.f) + t;
}
__device__ __forceinline__ float sg_(float x) {
  return frcp_(1.f + fexp2_(-x * L2E));
}
__device__ __forceinline__ void sm3_(float a, float b, float c,
                                     float& o0, float& o1, float& o2) {
  float m = fmaxf(a, fmaxf(b, c));
  float e0 = fexp2_((a - m) * L2E), e1 = fexp2_((b - m) * L2E), e2 = fexp2_((c - m) * L2E);
  float r = frcp_(e0 + e1 + e2);
  o0 = e0 * r; o1 = e1 * r; o2 = e2 * r;
}
__device__ __forceinline__ void sm5_(float a, float b, float c, float d, float e,
                                     float& o0, float& o1, float& o2, float& o3, float& o4) {
  float m = fmaxf(fmaxf(a, b), fmaxf(fmaxf(c, d), e));
  float e0 = fexp2_((a - m) * L2E), e1 = fexp2_((b - m) * L2E), e2 = fexp2_((c - m) * L2E),
        e3 = fexp2_((d - m) * L2E), e4 = fexp2_((e - m) * L2E);
  float r = frcp_(e0 + e1 + e2 + e3 + e4);
  o0 = e0 * r; o1 = e1 * r; o2 = e2 * r; o3 = e3 * r; o4 = e4 * r;
}

__device__ __forceinline__ v16h ld16_(const _Float16* p0, const _Float16* p1) {
  v8h a = *(const v8h*)p0;
  v8h b = *(const v8h*)p1;
  return __builtin_shufflevector(a, b, 0,1,2,3,4,5,6,7,8,9,10,11,12,13,14,15);
}
__device__ __forceinline__ v8f wmma_(v16h a, v16h b, v8f c) {
  return __builtin_amdgcn_wmma_f32_16x16x32_f16(false, a, false, b, (short)0, c, false, false);
}

// 32-row tile x 128x128 layer: shared B fragments feed two M-halves
__device__ __forceinline__ void mlp_layer32(_Float16* actw, const _Float16* wT,
                                            const float* bias, int lane) {
  const int m = lane & 15, hi = lane >> 4;
  v16h Alo[4], Ahi[4];
#pragma unroll
  for (int c = 0; c < 4; ++c) {
    Alo[c] = ld16_(actw + m*WSTR + 32*c + hi*8,        actw + m*WSTR + 32*c + 16 + hi*8);
    Ahi[c] = ld16_(actw + (m+16)*WSTR + 32*c + hi*8,   actw + (m+16)*WSTR + 32*c + 16 + hi*8);
  }
#pragma unroll
  for (int t = 0; t < 8; ++t) {
    int col = t*16 + m;
    float bv = bias[col];
    v8f alo = {bv,bv,bv,bv,bv,bv,bv,bv};
    v8f ahi = {bv,bv,bv,bv,bv,bv,bv,bv};
#pragma unroll
    for (int c = 0; c < 4; ++c) {
      v16h B = ld16_(wT + col*WSTR + 32*c + hi*8, wT + col*WSTR + 32*c + 16 + hi*8);
      alo = wmma_(Alo[c], B, alo);
      ahi = wmma_(Ahi[c], B, ahi);
    }
#pragma unroll
    for (int i = 0; i < 8; ++i) {
      actw[(i + hi*8)*WSTR + col]        = (_Float16)sp_(alo[i]);
      actw[(i + hi*8 + 16)*WSTR + col]   = (_Float16)sp_(ahi[i]);
    }
  }
}

// Final layer 128->64 (59 used), f32 logits written into the SAME LDS region (alias):
// all A fragments are read into registers before the first store; DS is in-order per wave.
__device__ __forceinline__ void mlp_layer3_32(_Float16* actw, float* lg,
                                              const _Float16* w3T, const float* bias, int lane) {
  const int m = lane & 15, hi = lane >> 4;
  v16h Alo[4], Ahi[4];
#pragma unroll
  for (int c = 0; c < 4; ++c) {
    Alo[c] = ld16_(actw + m*WSTR + 32*c + hi*8,        actw + m*WSTR + 32*c + 16 + hi*8);
    Ahi[c] = ld16_(actw + (m+16)*WSTR + 32*c + hi*8,   actw + (m+16)*WSTR + 32*c + 16 + hi*8);
  }
#pragma unroll
  for (int t = 0; t < 4; ++t) {
    int col = t*16 + m;
    float bv = bias[col];
    v8f alo = {bv,bv,bv,bv,bv,bv,bv,bv};
    v8f ahi = {bv,bv,bv,bv,bv,bv,bv,bv};
#pragma unroll
    for (int c = 0; c < 4; ++c) {
      v16h B = ld16_(w3T + col*WSTR + 32*c + hi*8, w3T + col*WSTR + 32*c + 16 + hi*8);
      alo = wmma_(Alo[c], B, alo);
      ahi = wmma_(Ahi[c], B, ahi);
    }
#pragma unroll
    for (int i = 0; i < 8; ++i) {
      lg[(i + hi*8)*LSTR + col]      = alo[i];
      lg[(i + hi*8 + 16)*LSTR + col] = ahi[i];
    }
  }
}

#define BR(par, ch) do { float a_ = p[par]; float s_ = sg_(l[ch]); \
                         p[ch] = a_*s_; p[par] = a_*(1.f-s_); } while(0)
#define ST3(par, gate, c0, c1, c2) do { float a_ = p[par]; float g_ = sg_(l[gate]); \
  float e0_,e1_,e2_; sm3_(l[c0],l[c1],l[c2],e0_,e1_,e2_); \
  p[c0]=a_*g_*e0_; p[c1]=a_*g_*e1_; p[c2]=a_*g_*e2_; p[par]=a_*(1.f-g_); } while(0)
#define ST5(par, gate, c0, c1, c2, c3, c4) do { float a_ = p[par]; float g_ = sg_(l[gate]); \
  float e0_,e1_,e2_,e3_,e4_; sm5_(l[c0],l[c1],l[c2],l[c3],l[c4],e0_,e1_,e2_,e3_,e4_); \
  p[c0]=a_*g_*e0_; p[c1]=a_*g_*e1_; p[c2]=a_*g_*e2_; p[c3]=a_*g_*e3_; p[c4]=a_*g_*e4_; \
  p[par]=a_*(1.f-g_); } while(0)

__global__ void __launch_bounds__(128)
skinning_field_kernel(const float* __restrict__ xyz, const float* __restrict__ quats,
                      const float* __restrict__ tfs,
                      const float* __restrict__ amin, const float* __restrict__ amax,
                      const float* __restrict__ W0, const float* __restrict__ b0,
                      const float* __restrict__ W1, const float* __restrict__ b1,
                      const float* __restrict__ W2, const float* __restrict__ b2,
                      const float* __restrict__ W3, const float* __restrict__ b3,
                      float* __restrict__ out) {
  extern __shared__ char smem[];
  _Float16* w1T  = (_Float16*)smem;              // 128*WSTR
  _Float16* w2T  = w1T + 128*WSTR;
  _Float16* w3T  = w2T + 128*WSTR;               // 64*WSTR
  _Float16* tfsB = w3T + 64*WSTR;                // 16*TSTR
  float* w0s = (float*)(tfsB + 16*TSTR);         // 3*128
  float* b0s = w0s + 384;
  float* b1s = b0s + 128;
  float* b2s = b1s + 128;
  float* b3s = b2s + 128;                        // 64
  float* nrm = b3s + 64;                         // 8 (scale[0..2], offset[4..6])
  _Float16* act = (_Float16*)(nrm + 8);          // 4 waves * 32*WSTR  (aliased by f32 logits)

  const int tid = threadIdx.x;

  // ---- stage weights (fp32 -> f16, transposed to out-feature-major) ----
  for (int idx = tid; idx < 128*128; idx += 128) {
    int k = idx >> 7, n = idx & 127;
    w1T[n*WSTR + k] = (_Float16)W1[idx];
    w2T[n*WSTR + k] = (_Float16)W2[idx];
  }
  for (int idx = tid; idx < 64*WSTR; idx += 128) w3T[idx] = (_Float16)0.f;
  for (int idx = tid; idx < 16*TSTR; idx += 128) tfsB[idx] = (_Float16)0.f;
  if (tid < 64) b3s[tid] = 0.f;
  __syncthreads();
  for (int idx = tid; idx < 128*59; idx += 128) {
    int k = idx / 59, n = idx % 59;
    w3T[n*WSTR + k] = (_Float16)W3[idx];
  }
  for (int idx = tid; idx < 55*16; idx += 128) {
    int k = idx >> 4, n = idx & 15;
    tfsB[n*TSTR + k] = (_Float16)tfs[idx];
  }
  for (int idx = tid; idx < 384; idx += 128) w0s[idx] = W0[idx];
  b0s[tid] = b0[tid];
  b1s[tid] = b1[tid];
  b2s[tid] = b2[tid];
  if (tid < 59) b3s[tid] = b3[tid];
  if (tid < 3) {
    float mn = amin[tid], mx = amax[tid];
    float s = 2.f / (mx - mn);
    nrm[tid] = s;
    nrm[4 + tid] = -mn * s - 1.f;
  }
  __syncthreads();

  const int wid  = tid >> 5;
  const int lane = tid & 31;
  const int m0   = lane & 15;
  const int hi   = lane >> 4;

  _Float16* Aw = act + wid * (32*WSTR);
  float*    Lw = (float*)Aw;     // alias: 272B row stride both ways

  float* ox   = out;
  float* orot = out + 3*NPTS;
  float* oq   = out + 12*NPTS;
  float* oT   = out + 16*NPTS;

  const int NT = NPTS / 32;
  const int gw = blockIdx.x * 4 + wid;
  const int stride = gridDim.x * 4;

  float sx = nrm[0], sy = nrm[1], sz = nrm[2];
  float cx = nrm[4], cy = nrm[5], cz = nrm[6];

  for (int tile = gw; tile < NT; tile += stride) {
    const int tb = tile * 32;
    const int pt = tb + lane;

    // prefetch next tile's point data into cache (global_prefetch_b8)
    {
      int nt = tile + stride;
      if (nt < NT) {
        __builtin_prefetch(xyz + (nt*32 + lane)*3, 0, 3);
        __builtin_prefetch(quats + (nt*32 + lane)*4, 0, 3);
      }
    }

    // ---- own point: load + normalize (kept in registers for the whole tile) ----
    float x = xyz[pt*3+0], y = xyz[pt*3+1], z = xyz[pt*3+2];
    float xn = fmaf(x, sx, cx), yn = fmaf(y, sy, cy), zn = fmaf(z, sz, cz);

    // ---- layer 0 (K=3) on VALU: each lane fills its own act row (128 f16) ----
#pragma unroll
    for (int g = 0; g < 16; ++g) {
      v8h vv;
#pragma unroll
      for (int e = 0; e < 8; ++e) {
        int f = g*8 + e;
        float a = b0s[f] + xn*w0s[f] + yn*w0s[128+f] + zn*w0s[256+f];
        vv[e] = (_Float16)sp_(a);
      }
      *(v8h*)(Aw + lane*WSTR + g*8) = vv;
    }

    // ---- layers 1..3 via WMMA (32 rows, shared B fragments) ----
    mlp_layer32(Aw, w1T, b1s, lane);
    mlp_layer32(Aw, w2T, b2s, lane);
    mlp_layer3_32(Aw, Lw, w3T, b3s, lane);

    // ---- hierarchical softmax: one point per lane, full wave ----
    {
      float l[59];
#pragma unroll
      for (int k = 0; k < 59; ++k) l[k] = Lw[lane*LSTR + k];
      float p[55];
      {
        float e0, e1, e2; sm3_(l[1], l[2], l[3], e0, e1, e2);
        float s0v = sg_(l[0]);
        p[1]=s0v*e0; p[2]=s0v*e1; p[3]=s0v*e2; p[0]=1.f-s0v;
      }
      BR(1,4);  BR(2,5);  BR(3,6);
      BR(4,7);  BR(5,8);  BR(6,9);
      BR(7,10); BR(8,11);
      ST3(9, 55, 12, 13, 14);
      BR(12,15);
      BR(13,16); BR(14,17);
      BR(16,18); BR(17,19);
      BR(18,20); BR(19,21);
      ST3(15, 56, 22, 23, 24);
      ST5(20, 57, 25, 28, 31, 34, 37);
      ST5(21, 58, 40, 43, 46, 49, 52);
      BR(25,26); BR(28,29); BR(31,32); BR(34,35); BR(37,38);
      BR(26,27); BR(29,30); BR(32,33); BR(35,36); BR(38,39);
      BR(40,41); BR(43,44); BR(46,47); BR(49,50); BR(52,53);
      BR(41,42); BR(44,45); BR(47,48); BR(50,51); BR(53,54);
#pragma unroll
      for (int k = 0; k < 55; ++k) Aw[lane*WSTR + k] = (_Float16)p[k];
#pragma unroll
      for (int k = 55; k < 64; ++k) Aw[lane*WSTR + k] = (_Float16)0.f;
    }

    // ---- blend: T_fwd = pts_W (32x64) @ tfs (64x16) via 4 WMMAs ----
    {
      v16h A0lo = ld16_(Aw + m0*WSTR +  0 + hi*8,      Aw + m0*WSTR + 16 + hi*8);
      v16h A1lo = ld16_(Aw + m0*WSTR + 32 + hi*8,      Aw + m0*WSTR + 48 + hi*8);
      v16h A0hi = ld16_(Aw + (m0+16)*WSTR +  0 + hi*8, Aw + (m0+16)*WSTR + 16 + hi*8);
      v16h A1hi = ld16_(Aw + (m0+16)*WSTR + 32 + hi*8, Aw + (m0+16)*WSTR + 48 + hi*8);
      v16h B0 = ld16_(tfsB + m0*TSTR +  0 + hi*8, tfsB + m0*TSTR + 16 + hi*8);
      v16h B1 = ld16_(tfsB + m0*TSTR + 32 + hi*8, tfsB + m0*TSTR + 48 + hi*8);
      v8f Tlo = {}, Thi = {};
      Tlo = wmma_(A0lo, B0, Tlo);
      Tlo = wmma_(A1lo, B1, Tlo);
      Thi = wmma_(A0hi, B0, Thi);
      Thi = wmma_(A1hi, B1, Thi);
#pragma unroll
      for (int i = 0; i < 8; ++i) {
        int r0 = i + hi*8, r1 = r0 + 16;
        oT[(tb + r0)*16 + m0] = Tlo[i];
        oT[(tb + r1)*16 + m0] = Thi[i];
        Lw[r0*LSTR + m0] = Tlo[i];
        Lw[r1*LSTR + m0] = Thi[i];
      }
    }

    // ---- per-point tail: x_bar, rot_bar, quat_bar (full wave, own point) ----
    {
      float T[16];
#pragma unroll
      for (int j = 0; j < 16; ++j) T[j] = Lw[lane*LSTR + j];
#pragma unroll
      for (int j = 0; j < 3; ++j)
        ox[pt*3+j] = T[j*4+0]*x + T[j*4+1]*y + T[j*4+2]*z + T[j*4+3];

      float q0 = quats[pt*4+0], q1 = quats[pt*4+1], q2 = quats[pt*4+2], q3 = quats[pt*4+3];
      float qn = frsq_(q0*q0 + q1*q1 + q2*q2 + q3*q3);
      float r = q0*qn, xq = q1*qn, yq = q2*qn, zq = q3*qn;
      float R00 = 1.f - 2.f*(yq*yq + zq*zq), R01 = 2.f*(xq*yq - r*zq), R02 = 2.f*(xq*zq + r*yq);
      float R10 = 2.f*(xq*yq + r*zq), R11 = 1.f - 2.f*(xq*xq + zq*zq), R12 = 2.f*(yq*zq - r*xq);
      float R20 = 2.f*(xq*zq - r*yq), R21 = 2.f*(yq*zq + r*xq), R22 = 1.f - 2.f*(xq*xq + yq*yq);

      float M[9];
      M[0] = T[0]*R00 + T[1]*R10 + T[2]*R20;
      M[1] = T[0]*R01 + T[1]*R11 + T[2]*R21;
      M[2] = T[0]*R02 + T[1]*R12 + T[2]*R22;
      M[3] = T[4]*R00 + T[5]*R10 + T[6]*R20;
      M[4] = T[4]*R01 + T[5]*R11 + T[6]*R21;
      M[5] = T[4]*R02 + T[5]*R12 + T[6]*R22;
      M[6] = T[8]*R00 + T[9]*R10 + T[10]*R20;
      M[7] = T[8]*R01 + T[9]*R11 + T[10]*R21;
      M[8] = T[8]*R02 + T[9]*R12 + T[10]*R22;
#pragma unroll
      for (int j = 0; j < 9; ++j) orot[pt*9+j] = M[j];

      float m00=M[0], m01=M[1], m02=M[2];
      float m10=M[3], m11=M[4], m12=M[5];
      float m20=M[6], m21=M[7], m22=M[8];
      float t0 = 1.f+m00+m11+m22, t1 = 1.f+m00-m11-m22;
      float t2 = 1.f-m00+m11-m22, t3 = 1.f-m00-m11+m22;
      int best = 0; float tbst = t0;
      if (t1 > tbst) { best = 1; tbst = t1; }
      if (t2 > tbst) { best = 2; tbst = t2; }
      if (t3 > tbst) { best = 3; tbst = t3; }
      float a0 = m21-m12, a1 = m02-m20, a2 = m10-m01;
      float b01 = m01+m10, b02 = m02+m20, b12 = m12+m21;
      float qa, qb, qc, qd;
      if (best == 0)      { qa = t0;  qb = a0;  qc = a1;  qd = a2; }
      else if (best == 1) { qa = a0;  qb = t1;  qc = b01; qd = b02; }
      else if (best == 2) { qa = a1;  qb = b01; qc = t2;  qd = b12; }
      else                { qa = a2;  qb = b02; qc = b12; qd = t3; }
      float sc = 0.5f * frsq_(fmaxf(tbst, 1e-8f));
      qa *= sc; qb *= sc; qc *= sc; qd *= sc;
      float inq = frsq_(qa*qa + qb*qb + qc*qc + qd*qd);
      oq[pt*4+0] = qa*inq; oq[pt*4+1] = qb*inq;
      oq[pt*4+2] = qc*inq; oq[pt*4+3] = qd*inq;
    }
  }
}

extern "C" void kernel_launch(void* const* d_in, const int* in_sizes, int n_in,
                              void* d_out, int out_size, void* d_ws, size_t ws_size,
                              hipStream_t stream) {
  const float* xyz   = (const float*)d_in[0];
  const float* quats = (const float*)d_in[1];
  const float* tfs   = (const float*)d_in[2];
  const float* amin  = (const float*)d_in[3];
  const float* amax  = (const float*)d_in[4];
  const float* W0    = (const float*)d_in[5];
  const float* b0    = (const float*)d_in[6];
  const float* W1    = (const float*)d_in[7];
  const float* b1    = (const float*)d_in[8];
  const float* W2    = (const float*)d_in[9];
  const float* b2    = (const float*)d_in[10];
  const float* W3    = (const float*)d_in[11];
  const float* b3    = (const float*)d_in[12];
  float* out = (float*)d_out;

  (void)hipFuncSetAttribute((const void*)skinning_field_kernel,
                            hipFuncAttributeMaxDynamicSharedMemorySize, SMEM_BYTES);

  dim3 grid(1024), block(128);
  skinning_field_kernel<<<grid, block, SMEM_BYTES, stream>>>(
      xyz, quats, tfs, amin, amax, W0, b0, W1, b1, W2, b2, W3, b3, out);
}
